// PhysicsInformedLoss_70325794505033
// MI455X (gfx1250) — compile-verified, matrix operand order checked by reading, not run on hardware
//
#include <hip/hip_runtime.h>
#include <hip/hip_bf16.h>

// Problem constants (match reference).
constexpr int kB = 32;
constexpr int kN = 1024;
constexpr int kT = 48;
constexpr int kC = kB * kT;                       // 1536 contraction columns
constexpr int kNSB = kN / 32;                     // 32 superblocks of 32 rows
constexpr int kTilesSB = kNSB * (kNSB + 1) / 2;   // 528 upper-tri superblock pairs
constexpr int kWavesPerBlock = 4;
constexpr float kLambdaPhys = 0.1f;
constexpr float kLambdaSmooth = 0.01f;

typedef __attribute__((ext_vector_type(16))) __bf16 v16bf;
typedef __attribute__((ext_vector_type(8)))  float  v8f;

// ---------------------------------------------------------------- init
__global__ void pil_init_kernel(float* accum) {
  if (threadIdx.x < 4) accum[threadIdx.x] = 0.0f;
}

// ------------------------------------------------- pred -> bf16 P, q_i
// Pb[i][c] = pred[b,i,t] with c = b*T+t (bf16).  q_i = sum_{b,t} pred^2.
__global__ void pil_convert_kernel(const float* __restrict__ pred,
                                   __bf16* __restrict__ Pb,
                                   float* __restrict__ qbuf) {
  const int i = blockIdx.x;
  float acc = 0.0f;
  for (int c = threadIdx.x; c < kC; c += blockDim.x) {
    const int b = c / kT;
    const int t = c - b * kT;
    const float x = pred[(size_t)b * kN * kT + (size_t)i * kT + t];
    Pb[(size_t)i * kC + c] = (__bf16)x;
    acc += x * x;
  }
  __shared__ float sm[256];
  sm[threadIdx.x] = acc;
  __syncthreads();
  for (int s = blockDim.x >> 1; s > 0; s >>= 1) {
    if (threadIdx.x < s) sm[threadIdx.x] += sm[threadIdx.x + s];
    __syncthreads();
  }
  if (threadIdx.x == 0) qbuf[i] = sm[0];
}

// --------------------------------------------------- r_i (sym rowsums)
// r_i = sum_{j>i} adj[i,j] + sum_{j<i} adj[j,i]  (adj already thresholded >=0)
__global__ void pil_rowsum_kernel(const float* __restrict__ adj,
                                  float* __restrict__ rbuf) {
  const int i = blockIdx.x;
  float acc = 0.0f;
  for (int j = threadIdx.x; j < kN; j += blockDim.x) {
    if (j > i)      acc += adj[(size_t)i * kN + j];
    else if (j < i) acc += adj[(size_t)j * kN + i];
  }
  __shared__ float sm[128];
  sm[threadIdx.x] = acc;
  __syncthreads();
  for (int s = blockDim.x >> 1; s > 0; s >>= 1) {
    if (threadIdx.x < s) sm[threadIdx.x] += sm[threadIdx.x + s];
    __syncthreads();
  }
  if (threadIdx.x == 0) rbuf[i] = sm[0];
}

// -------------------------------------- fused pred-MSE + physics-MSE
__global__ void pil_mse_kernel(const float* __restrict__ pred,
                               const float* __restrict__ targ,
                               const float* __restrict__ phys,
                               float* __restrict__ accum) {
  const size_t total = (size_t)kB * kN * kT;
  size_t idx = (size_t)blockIdx.x * blockDim.x + threadIdx.x;
  const size_t stride = (size_t)gridDim.x * blockDim.x;
  float s1 = 0.0f, s2 = 0.0f;
  for (; idx < total; idx += stride) {
    const float d = pred[idx] - targ[idx];
    s1 += d * d;
    const float r = phys[idx];
    s2 += r * r;
  }
  __shared__ float sm1[256], sm2[256];
  sm1[threadIdx.x] = s1;
  sm2[threadIdx.x] = s2;
  __syncthreads();
  for (int s = blockDim.x >> 1; s > 0; s >>= 1) {
    if (threadIdx.x < s) {
      sm1[threadIdx.x] += sm1[threadIdx.x + s];
      sm2[threadIdx.x] += sm2[threadIdx.x + s];
    }
    __syncthreads();
  }
  if (threadIdx.x == 0) {
    atomicAdd(&accum[0], sm1[0]);
    atomicAdd(&accum[1], sm2[0]);
  }
}

// ---------------------------------------- weighted Gram via WMMA bf16
// One wave computes a 32x32 superblock of Gm = P P^T as 2x2 tiles of
// v_wmma_f32_16x16x32_bf16 accumulators, contracts against W_sym on the
// fly, and atomically accumulates  sum_{i!=j} W_sym[i,j] * Gm[i,j].
__global__ void __launch_bounds__(kWavesPerBlock * 32)
pil_smooth_gemm_kernel(const __bf16* __restrict__ Pb,
                       const float* __restrict__ adj,
                       float* __restrict__ accum) {
  const int lane = threadIdx.x & 31;
  const int wave = threadIdx.x >> 5;
  const int q = blockIdx.x * kWavesPerBlock + wave;
  if (q >= kTilesSB) return;   // whole wave exits: active waves keep EXEC all-1s

  // Decode upper-triangular superblock pair (Isb <= Jsb).
  int Isb = 0, rem = q;
  while (rem >= kNSB - Isb) { rem -= kNSB - Isb; ++Isb; }
  const int Jsb = Isb + rem;

  // A(16x32 bf16) and B(32x16 bf16) fragments are the same gather:
  // lane reads 16 consecutive bf16 from row base+(lane&15),
  // K-offset (lane>>4)*16 within each 32-wide K block.
  const int rsel = lane & 15;
  const int ksel = (lane >> 4) << 4;

  const __bf16* a0 = Pb + (size_t)(Isb * 32 + rsel) * kC + ksel;
  const __bf16* a1 = a0 + (size_t)16 * kC;
  const __bf16* b0 = Pb + (size_t)(Jsb * 32 + rsel) * kC + ksel;
  const __bf16* b1 = b0 + (size_t)16 * kC;

  v8f c00 = {}, c01 = {}, c10 = {}, c11 = {};
#pragma unroll 2
  for (int kk = 0; kk < kC / 32; ++kk) {           // 48 K-steps
    const v16bf fa0 = *(const v16bf*)(a0 + kk * 32);
    const v16bf fa1 = *(const v16bf*)(a1 + kk * 32);
    const v16bf fb0 = *(const v16bf*)(b0 + kk * 32);
    const v16bf fb1 = *(const v16bf*)(b1 + kk * 32);
    c00 = __builtin_amdgcn_wmma_f32_16x16x32_bf16(false, fa0, false, fb0,
                                                  (short)0, c00, false, false);
    c01 = __builtin_amdgcn_wmma_f32_16x16x32_bf16(false, fa0, false, fb1,
                                                  (short)0, c01, false, false);
    c10 = __builtin_amdgcn_wmma_f32_16x16x32_bf16(false, fa1, false, fb0,
                                                  (short)0, c10, false, false);
    c11 = __builtin_amdgcn_wmma_f32_16x16x32_bf16(false, fa1, false, fb1,
                                                  (short)0, c11, false, false);
  }

  // C/D layout: VGPR r of lane L -> (m = r + 8*(L>>4), n = L&15).
  const int mbase = (lane >> 4) << 3;
  const int n = lane & 15;
  float partial = 0.0f;
#pragma unroll
  for (int y = 0; y < 2; ++y) {
    const int j = Jsb * 32 + y * 16 + n;
#pragma unroll
    for (int x = 0; x < 2; ++x) {
      const v8f cc = (x == 0) ? (y == 0 ? c00 : c01) : (y == 0 ? c10 : c11);
#pragma unroll
      for (int r = 0; r < 8; ++r) {
        const int i = Isb * 32 + x * 16 + mbase + r;
        float w = 0.0f;
        if (i < j)      w = adj[(size_t)i * kN + j];   // W_sym upper
        else if (i > j) w = adj[(size_t)j * kN + i];   // W_sym lower (mirror)
        partial += w * cc[r];
      }
    }
  }
  if (Isb != Jsb) partial *= 2.0f;  // mirrored superblock contributes equally

  for (int off = 16; off > 0; off >>= 1)
    partial += __shfl_down(partial, off, 32);
  if (lane == 0) atomicAdd(&accum[2], partial);
}

// ------------------------------------------------------------ finalize
__global__ void pil_finalize_kernel(const float* __restrict__ rbuf,
                                    const float* __restrict__ qbuf,
                                    const float* __restrict__ accum,
                                    float* __restrict__ out) {
  __shared__ float sm[1024];
  const int i = threadIdx.x;
  sm[i] = rbuf[i] * qbuf[i];
  __syncthreads();
  for (int s = 512; s > 0; s >>= 1) {
    if (i < s) sm[i] += sm[i + s];
    __syncthreads();
  }
  if (i == 0) {
    const float tot = (float)kB * (float)kN * (float)kT;
    const float pred_loss = accum[0] / tot;
    const float phys_loss = accum[1] / tot;
    const float term1 = sm[0] / (float)kB;            // sum_t sum_i r_i s_it
    const float term2 = -accum[2] / (float)kB;        // -(1/B) sum W_sym.Gm
    const float smooth_loss = (term1 + term2) / ((float)kN * (float)kT);
    out[0] = pred_loss + kLambdaPhys * phys_loss + kLambdaSmooth * smooth_loss;
    out[1] = pred_loss;
    out[2] = phys_loss;
    out[3] = smooth_loss;
  }
}

// -------------------------------------------------------------- launch
extern "C" void kernel_launch(void* const* d_in, const int* in_sizes, int n_in,
                              void* d_out, int out_size, void* d_ws, size_t ws_size,
                              hipStream_t stream) {
  const float* pred = (const float*)d_in[0];
  const float* targ = (const float*)d_in[1];
  const float* phys = (const float*)d_in[2];
  const float* adj  = (const float*)d_in[3];
  float* out = (float*)d_out;

  // Workspace: bf16 P (3 MB) | r (4 KB) | q (4 KB) | accum (16 B)
  __bf16* Pb   = (__bf16*)d_ws;
  float*  rbuf = (float*)((char*)d_ws + (size_t)kN * kC * sizeof(__bf16));
  float*  qbuf = rbuf + kN;
  float*  accum = qbuf + kN;

  pil_init_kernel<<<1, 32, 0, stream>>>(accum);
  pil_convert_kernel<<<kN, 256, 0, stream>>>(pred, Pb, qbuf);
  pil_rowsum_kernel<<<kN, 128, 0, stream>>>(adj, rbuf);
  pil_mse_kernel<<<1024, 256, 0, stream>>>(pred, targ, phys, accum);
  pil_smooth_gemm_kernel<<<kTilesSB / kWavesPerBlock, kWavesPerBlock * 32, 0,
                           stream>>>(Pb, adj, accum);
  pil_finalize_kernel<<<1, 1024, 0, stream>>>(rbuf, qbuf, accum, out);
}
